// ResidualGNNLayer_7267084664911
// MI455X (gfx1250) — matrix operand order repaired and checked by reading, not compiled, not run on hardware
//
#include <hip/hip_runtime.h>

typedef __attribute__((ext_vector_type(16))) __bf16 v16bf;
typedef __attribute__((ext_vector_type(8)))  __bf16 v8bf;
typedef __attribute__((ext_vector_type(8)))  float  v8f;

#define DF 256          // feature dim
#define GEMM_ROWS 32    // rows per block in GEMM
#define ALPHA 0.5f
#define EPS 1e-5f

// ---------------- degree / norm ----------------
__global__ __launch_bounds__(256) void k_zero(float* __restrict__ p, int n) {
    int i = blockIdx.x * 256 + threadIdx.x;
    if (i < n) p[i] = 0.0f;
}

__global__ __launch_bounds__(256) void k_deg(const int* __restrict__ dst,
                                             const float* __restrict__ ew,
                                             float* __restrict__ deg, int E) {
    int e = blockIdx.x * 256 + threadIdx.x;
    if (e < E) atomicAdd(deg + dst[e], ew[e]);
}

__global__ __launch_bounds__(256) void k_dinv(float* __restrict__ deg, int N) {
    int i = blockIdx.x * 256 + threadIdx.x;
    if (i < N) deg[i] = rsqrtf(deg[i] + 1.0f);   // becomes dinv in place
}

// ---------------- W -> bf16, transposed (column-major) ----------------
// Wt[c*256 + k] = bf16(W[k*256 + c]) so a B fragment (16 consecutive K of one
// column) is a contiguous 32-byte load per lane.
__global__ __launch_bounds__(256) void k_wprep(const float* __restrict__ W,
                                               __bf16* __restrict__ Wt) {
    int idx = blockIdx.x * 256 + threadIdx.x;  // 65536 total
    int k = idx >> 8;
    int c = idx & 255;
    Wt[c * 256 + k] = (__bf16)W[k * 256 + c];
}

// ---------------- h = x @ W via v_wmma_f32_16x16x32_bf16 ----------------
__global__ __launch_bounds__(256) void k_gemm(const float* __restrict__ x,
                                              const __bf16* __restrict__ Wt,
                                              float* __restrict__ h, int N) {
    __shared__ __bf16 lx[GEMM_ROWS][DF];   // 16 KB, 32 rows of x in bf16
    const int tid  = threadIdx.x;
    const int row0 = blockIdx.x * GEMM_ROWS;
    const float4* x4 = (const float4*)x;

    // Stage 32x256 fp32 -> bf16 LDS (2048 float4, 8 per thread)
    #pragma unroll
    for (int i = 0; i < 8; ++i) {
        int idx = tid + i * 256;
        int r   = idx >> 6;          // 64 float4 per row
        int c4  = idx & 63;
        int row = row0 + r;
        float4 v = make_float4(0.f, 0.f, 0.f, 0.f);
        if (row < N) v = x4[(size_t)row * 64 + c4];
        int c = c4 * 4;
        lx[r][c + 0] = (__bf16)v.x;
        lx[r][c + 1] = (__bf16)v.y;
        lx[r][c + 2] = (__bf16)v.z;
        lx[r][c + 3] = (__bf16)v.w;
    }
    __syncthreads();

    const int wave = tid >> 5;       // 8 waves
    const int lane = tid & 31;
    const int lr   = lane & 15;      // row (A) / col (B,C,D) within tile
    const int hi   = lane >> 4;      // half-wave selector
    const int rt   = wave >> 2;      // row tile 0..1
    const int ct0  = (wave & 3) * 4; // first of 4 col tiles
    const int arow = rt * 16 + lr;

    v8f acc[4];
    #pragma unroll
    for (int t = 0; t < 4; ++t) acc[t] = 0.0f;

    #pragma unroll
    for (int ks = 0; ks < 8; ++ks) {
        const int k0 = ks * 32;
        // A fragment (16x32 bf16): lane half 'hi' holds K run [hi*16, hi*16+16)
        // of row 'arow' (contiguous order, matching B)
        const int ko = k0 + hi * 16;
        v8bf alo = *(const v8bf*)&lx[arow][ko];
        v8bf ahi = *(const v8bf*)&lx[arow][ko + 8];
        v16bf a;
        #pragma unroll
        for (int i = 0; i < 8; ++i) { a[i] = alo[i]; a[i + 8] = ahi[i]; }

        #pragma unroll
        for (int t = 0; t < 4; ++t) {
            // B fragment (32x16 bf16): lane = column, halves hold K+0..15 / K+16..31
            const int col = (ct0 + t) * 16 + lr;
            v16bf b = *(const v16bf*)(Wt + (size_t)col * 256 + k0 + hi * 16);
            acc[t] = __builtin_amdgcn_wmma_f32_16x16x32_bf16(
                false, a, false, b, (short)0, acc[t], false, false);
        }
    }

    // C/D layout: VGPR j -> M = j + 8*hi, N = lr
    #pragma unroll
    for (int t = 0; t < 4; ++t) {
        const int col = (ct0 + t) * 16 + lr;
        #pragma unroll
        for (int j = 0; j < 8; ++j) {
            int row = row0 + rt * 16 + j + 8 * hi;
            if (row < N) h[(size_t)row * DF + col] = acc[t][j];
        }
    }
}

// ---------------- agg init: self-loop term (writes d_out) ----------------
__global__ __launch_bounds__(256) void k_selfloop(const float* __restrict__ h,
                                                  const float* __restrict__ dinv,
                                                  float* __restrict__ agg,
                                                  long long nvec4) {
    long long idx = (long long)blockIdx.x * 256 + threadIdx.x;  // over N*64 float4
    if (idx >= nvec4) return;
    int node = (int)(idx >> 6);
    float w = dinv[node];
    w = w * w;
    float4 v = ((const float4*)h)[idx];
    ((float4*)agg)[idx] = make_float4(w * v.x, w * v.y, w * v.z, w * v.w);
}

// ---------------- edge scatter: one wave per edge ----------------
__global__ __launch_bounds__(256) void k_scatter(const int* __restrict__ src,
                                                 const int* __restrict__ dst,
                                                 const float* __restrict__ ew,
                                                 const float* __restrict__ dinv,
                                                 const float* __restrict__ h,
                                                 float* __restrict__ agg, int E) {
    int gw   = (blockIdx.x * 256 + threadIdx.x) >> 5;
    int lane = threadIdx.x & 31;
    if (gw >= E) return;
    int s = src[gw], d = dst[gw];
    float nrm = dinv[s] * ew[gw] * dinv[d];
    const float4* hs = (const float4*)(h + (size_t)s * DF);
    float* ad = agg + (size_t)d * DF;
    #pragma unroll
    for (int i = 0; i < 2; ++i) {
        int f4 = lane + i * 32;
        float4 v = hs[f4];
        atomicAdd(ad + f4 * 4 + 0, nrm * v.x);
        atomicAdd(ad + f4 * 4 + 1, nrm * v.y);
        atomicAdd(ad + f4 * 4 + 2, nrm * v.z);
        atomicAdd(ad + f4 * 4 + 3, nrm * v.w);
    }
}

// ---------------- bias + LayerNorm + residual + ReLU (in place) -------------
__global__ __launch_bounds__(256) void k_final(const float* __restrict__ x,
                                               const float* __restrict__ bias,
                                               const float* __restrict__ gamma,
                                               const float* __restrict__ beta,
                                               float* __restrict__ out, int N) {
    int node = (blockIdx.x * 256 + threadIdx.x) >> 5;   // one wave32 per node
    int lane = threadIdx.x & 31;
    if (node >= N) return;
    float* row = out + (size_t)node * DF;
    const float* xr = x + (size_t)node * DF;

    float v[8];
    float s = 0.0f;
    #pragma unroll
    for (int i = 0; i < 8; ++i) {
        int c = lane + i * 32;
        v[i] = row[c] + bias[c];
        s += v[i];
    }
    #pragma unroll
    for (int off = 16; off >= 1; off >>= 1) s += __shfl_xor(s, off, 32);
    float mu = s * (1.0f / DF);

    float q = 0.0f;
    #pragma unroll
    for (int i = 0; i < 8; ++i) { float d = v[i] - mu; q += d * d; }
    #pragma unroll
    for (int off = 16; off >= 1; off >>= 1) q += __shfl_xor(q, off, 32);
    float inv = rsqrtf(q * (1.0f / DF) + EPS);

    #pragma unroll
    for (int i = 0; i < 8; ++i) {
        int c = lane + i * 32;
        float hn = (v[i] - mu) * inv * gamma[c] + beta[c];
        float r  = ALPHA * hn + (1.0f - ALPHA) * xr[c];
        row[c] = fmaxf(r, 0.0f);
    }
}

extern "C" void kernel_launch(void* const* d_in, const int* in_sizes, int n_in,
                              void* d_out, int out_size, void* d_ws, size_t ws_size,
                              hipStream_t stream) {
    const float* x     = (const float*)d_in[0];
    const int*   ei    = (const int*)d_in[1];
    const float* ew    = (const float*)d_in[2];
    const float* W     = (const float*)d_in[3];
    const float* bias  = (const float*)d_in[4];
    const float* gamma = (const float*)d_in[5];
    const float* beta  = (const float*)d_in[6];

    const int N = in_sizes[0] / DF;
    const int E = in_sizes[2];
    const int* src = ei;          // edge_index[0]
    const int* dst = ei + E;      // edge_index[1]
    float* out = (float*)d_out;   // doubles as agg buffer

    // workspace layout: deg/dinv | Wt(bf16) | h
    char* ws = (char*)d_ws;
    float*  deg = (float*)ws;
    size_t  off = (((size_t)N * 4) + 255) & ~(size_t)255;
    __bf16* Wt  = (__bf16*)(ws + off);
    off += (((size_t)DF * DF * 2) + 255) & ~(size_t)255;
    float*  h   = (float*)(ws + off);

    k_zero<<<(N + 255) / 256, 256, 0, stream>>>(deg, N);
    k_deg<<<(E + 255) / 256, 256, 0, stream>>>(dst, ew, deg, E);
    k_dinv<<<(N + 255) / 256, 256, 0, stream>>>(deg, N);
    k_wprep<<<(DF * DF) / 256, 256, 0, stream>>>(W, Wt);
    k_gemm<<<(N + GEMM_ROWS - 1) / GEMM_ROWS, 256, 0, stream>>>(x, Wt, h, N);

    long long nvec4 = (long long)N * (DF / 4);
    k_selfloop<<<(unsigned)((nvec4 + 255) / 256), 256, 0, stream>>>(h, deg, out, nvec4);
    k_scatter<<<(E + 7) / 8, 256, 0, stream>>>(src, dst, ew, deg, h, out, E);
    k_final<<<(N + 7) / 8, 256, 0, stream>>>(x, bias, gamma, beta, out, N);
}
